// VectorQuantiser_73358041416270
// MI455X (gfx1250) — matrix operand order repaired and checked by reading, not compiled, not run on hardware
//
#include <hip/hip_runtime.h>
#include <hip/hip_bf16.h>

typedef __attribute__((ext_vector_type(16))) __bf16 v16bf;
typedef __attribute__((ext_vector_type(8)))  float  v8f;

#define N_CODES   512
#define CODE_DIM  32
#define ROWS      (8192 * 64)                 // 524288 vectors
#define ZQ_ELEMS  ((size_t)ROWS * CODE_DIM)   // 16777216
#define IDX_OFF   (ZQ_ELEMS)
#define LOSS_OFF  (ZQ_ELEMS + (size_t)ROWS)

static __device__ __forceinline__ unsigned short f2bf(float f) {
    unsigned int u = __float_as_uint(f);
    u += 0x7FFFu + ((u >> 16) & 1u);          // round-to-nearest-even
    return (unsigned short)(u >> 16);
}

// ---------------------------------------------------------------------------
// Prep: fp32 codebook -> bf16 copy + per-code squared norms; zero loss accum.
// ---------------------------------------------------------------------------
__global__ void vq_prep_kernel(const float* __restrict__ cb,
                               float* __restrict__ ws_wn,
                               unsigned short* __restrict__ ws_cb,
                               float* __restrict__ accum) {
    int r = blockIdx.x * blockDim.x + threadIdx.x;
    if (r == 0) *accum = 0.0f;
    if (r < N_CODES) {
        const float* row = cb + r * CODE_DIM;
        float s = 0.0f;
#pragma unroll
        for (int i = 0; i < CODE_DIM; ++i) {
            float v = row[i];
            s += v * v;
            ws_cb[r * CODE_DIM + i] = f2bf(v);
        }
        ws_wn[r] = s;
    }
}

// ---------------------------------------------------------------------------
// Main: per wave, 16 rows x 512 codes via 32x V_WMMA_F32_16X16X32_BF16.
// Argmin via packed sort-key (distance mantissa low 9 bits = code index) so
// the inner loop is fma + and_or + min_num per slot, and the cross-lane
// reduction is pure shfl+min. z_q gather / loss use the exact fp32 codebook.
// ---------------------------------------------------------------------------
__global__ __launch_bounds__(256) void vq_main_kernel(
    const float* __restrict__ z_e,
    const float* __restrict__ cb_f32,
    const unsigned short* __restrict__ ws_cb,
    const float* __restrict__ ws_wn,
    float* __restrict__ out,
    float* __restrict__ accum)
{
    __shared__ __align__(32) unsigned short s_cb[N_CODES * CODE_DIM]; // 32 KB bf16
    __shared__ __align__(16) float s_wn[N_CODES];                     // 2 KB

    const int tid = threadIdx.x;

    // Cooperative LDS fill: 2048 uint4 of codebook + 128 float4 of norms.
    {
        const uint4* src = (const uint4*)ws_cb;
        uint4* dst = (uint4*)s_cb;
#pragma unroll
        for (int i = 0; i < 8; ++i) dst[tid + 256 * i] = src[tid + 256 * i];
        if (tid < 128) ((float4*)s_wn)[tid] = ((const float4*)ws_wn)[tid];
    }
    __syncthreads();

    const int lane  = tid & 31;
    const int wv    = tid >> 5;                     // 8 waves per block
    const int wrow0 = blockIdx.x * 128 + wv * 16;   // first of this wave's 16 rows
    const int nlo   = lane & 15;
    const int khalf = lane >> 4;                    // selects K half (contiguous K order)

    // ---- A fragment: 16x32 bf16, row M = nlo, K = khalf*16 + s (contiguous) ----
    v16bf a;
    {
        const float4* z4 = (const float4*)(z_e + (size_t)(wrow0 + nlo) * CODE_DIM);
        float4 p0 = z4[khalf * 4 + 0];   // K = khalf*16 + 0..3
        float4 p1 = z4[khalf * 4 + 1];   // K = khalf*16 + 4..7
        float4 p2 = z4[khalf * 4 + 2];   // K = khalf*16 + 8..11
        float4 p3 = z4[khalf * 4 + 3];   // K = khalf*16 + 12..15
        a[0]  = (__bf16)p0.x; a[1]  = (__bf16)p0.y; a[2]  = (__bf16)p0.z; a[3]  = (__bf16)p0.w;
        a[4]  = (__bf16)p1.x; a[5]  = (__bf16)p1.y; a[6]  = (__bf16)p1.z; a[7]  = (__bf16)p1.w;
        a[8]  = (__bf16)p2.x; a[9]  = (__bf16)p2.y; a[10] = (__bf16)p2.z; a[11] = (__bf16)p2.w;
        a[12] = (__bf16)p3.x; a[13] = (__bf16)p3.y; a[14] = (__bf16)p3.z; a[15] = (__bf16)p3.w;
    }

    // Packed keys: bits = (dist & ~0x1FF) | code. min_num over keys == argmin.
    float bestk[8];
#pragma unroll
    for (int s = 0; s < 8; ++s) bestk[s] = __builtin_inff();

#pragma unroll 4
    for (int t = 0; t < 32; ++t) {
        const int code = t * 16 + nlo;
        // B fragment: 32x16 bf16; lane n = nlo, K = khalf*16 .. +15 contiguous
        const v16bf b = *(const v16bf*)&s_cb[code * CODE_DIM + khalf * 16];
        v8f c = {};
        c = __builtin_amdgcn_wmma_f32_16x16x32_bf16(
                false, a, false, b, (short)0, c, false, false);
        const float wn = s_wn[code];
#pragma unroll
        for (int s = 0; s < 8; ++s) {
            // C layout: lanes 0-15 -> M = s, lanes 16-31 -> M = s + 8
            float dv = __builtin_fmaf(-2.0f, c[s], wn);
            unsigned key = (__float_as_uint(dv) & 0xFFFFFE00u) | (unsigned)code;
            bestk[s] = fminf(bestk[s], __uint_as_float(key));
        }
    }

    // Argmin across the 16 lanes of each half-wave: pure shuffle + min.
#pragma unroll
    for (int m = 8; m >= 1; m >>= 1) {
#pragma unroll
        for (int s = 0; s < 8; ++s)
            bestk[s] = fminf(bestk[s], __shfl_xor(bestk[s], m, 32));
    }

    int besti[8];
#pragma unroll
    for (int s = 0; s < 8; ++s) besti[s] = (int)(__float_as_uint(bestk[s]) & 511u);

    // ---- indices output (as float, concatenated after z_q) ----
    float* out_idx = out + IDX_OFF;
    if (nlo == 0) {
        const int base = wrow0 + khalf * 8;
#pragma unroll
        for (int s = 0; s < 8; ++s) out_idx[base + s] = (float)besti[s];
    }

    // Broadcast all 16 row winners to every lane.
    int idxs[16];
#pragma unroll
    for (int s = 0; s < 8; ++s) {
        idxs[s]     = __shfl(besti[s], 0, 32);   // rows 0..7
        idxs[8 + s] = __shfl(besti[s], 16, 32);  // rows 8..15
    }

    // ---- z_q gather (exact fp32 codebook) + loss partial ----
    float lsum = 0.0f;
#pragma unroll
    for (int m = 0; m < 16; ++m) {
        const int row = wrow0 + m;
        float q  = cb_f32[idxs[m] * CODE_DIM + lane];        // 32 lanes = 32 elems
        float ze = z_e[(size_t)row * CODE_DIM + lane];
        out[(size_t)row * CODE_DIM + lane] = q;              // z_q_st == z_q
        float d = q - ze;
        lsum += d * d;
    }
#pragma unroll
    for (int m = 16; m >= 1; m >>= 1) lsum += __shfl_xor(lsum, m, 32);
    if (lane == 0) atomicAdd(accum, lsum);
}

// ---------------------------------------------------------------------------
// Finalize: loss = (1 + BETA) * sum / numel   (both MSE terms are equal).
// ---------------------------------------------------------------------------
__global__ void vq_fin_kernel(const float* __restrict__ accum,
                              float* __restrict__ out) {
    out[LOSS_OFF] = accum[0] * 1.25f / (float)ZQ_ELEMS;
}

extern "C" void kernel_launch(void* const* d_in, const int* in_sizes, int n_in,
                              void* d_out, int out_size, void* d_ws, size_t ws_size,
                              hipStream_t stream) {
    const float* z_e = (const float*)d_in[0];
    const float* cb  = (const float*)d_in[1];
    float* out  = (float*)d_out;
    float* ws_f = (float*)d_ws;
    float* ws_wn = ws_f;                                    // 512 floats
    float* accum = ws_f + 512;                              // 1 float
    unsigned short* ws_cb = (unsigned short*)(ws_f + 1024); // 512*32 bf16 = 32 KB

    vq_prep_kernel<<<2, 256, 0, stream>>>(cb, ws_wn, ws_cb, accum);
    vq_main_kernel<<<4096, 256, 0, stream>>>(z_e, cb, ws_cb, ws_wn, out, accum);
    vq_fin_kernel<<<1, 1, 0, stream>>>(accum, out);
}
